// EncoderNetwork_7232724927123
// MI455X (gfx1250) — compile-verified, matrix operand order checked by reading, not run on hardware
//
#include <hip/hip_runtime.h>
#include <hip/hip_bf16.h>

// EncoderNetwork for MI455X (gfx1250, wave32, WMMA).
// All GEMMs use v_wmma_f32_16x16x32_f16 (f16 A/B, f32 accumulate).
// Workload is HBM-bandwidth bound (~3-4 GB moved vs ~21 GFLOP): one wave per
// 16-row tile, A fragments loaded as float4 (b128), B fragments pre-swizzled
// into WMMA lane order so each load is a contiguous v16h (2x b128).

#define Nn     500000
#define Ee     2000000
#define Ff     8
#define Hh     64
#define DEMB   32
#define Ll     8
#define Dseg   4096
#define Oseg   128

typedef __attribute__((ext_vector_type(16))) _Float16 v16h;
typedef __attribute__((ext_vector_type(8)))  _Float16 v8h;
typedef __attribute__((ext_vector_type(8)))  float    v8f;

__device__ __forceinline__ int kmap(int e, int c) {
    // 16-bit A/B fragment K index for element e (0..15), half c = lane/16.
    return (e < 8) ? (8 * c + e) : (16 + 8 * c + (e - 8));
}

__device__ __forceinline__ v8f wmma16(v16h a, v16h b, v8f c) {
    return __builtin_amdgcn_wmma_f32_16x16x32_f16(false, a, false, b,
                                                  (short)0, c, false, false);
}

__device__ __forceinline__ float lrelu(float v) {
    return v >= 0.0f ? v : 0.2f * v;
}

// Pack two contiguous float4 K-runs into a v16h A fragment.
__device__ __forceinline__ v16h packA(float4 p0, float4 p1, float4 p2, float4 p3) {
    v16h a;
    a[0]  = (_Float16)p0.x; a[1]  = (_Float16)p0.y; a[2]  = (_Float16)p0.z; a[3]  = (_Float16)p0.w;
    a[4]  = (_Float16)p1.x; a[5]  = (_Float16)p1.y; a[6]  = (_Float16)p1.z; a[7]  = (_Float16)p1.w;
    a[8]  = (_Float16)p2.x; a[9]  = (_Float16)p2.y; a[10] = (_Float16)p2.z; a[11] = (_Float16)p2.w;
    a[12] = (_Float16)p3.x; a[13] = (_Float16)p3.y; a[14] = (_Float16)p3.z; a[15] = (_Float16)p3.w;
    return a;
}

// ---------------------------------------------------------------------------
// Weight -> WMMA B-fragment panels, f16, K zero-padded.
// out[((kc*nT + t)*32 + lane)*16 + e] = W[(kc*32 + kmap(e, lane>>4)) * ncol
//                                         + t*16 + (lane&15)]  (0 if k>=ksrc)
__global__ void convert_frag_kernel(const float* __restrict__ W,
                                    _Float16* __restrict__ out,
                                    int ksrc, int nkc, int ncol) {
    int tid = blockIdx.x * blockDim.x + threadIdx.x;
    int nT = ncol >> 4;
    if (tid >= nkc * nT * 512) return;
    int e = tid & 15;
    int lane = (tid >> 4) & 31;
    int t = (tid >> 9) % nT;
    int kc = (tid >> 9) / nT;
    int c = lane >> 4, n = lane & 15;
    int k = kc * 32 + kmap(e, c);
    out[tid] = (k < ksrc) ? (_Float16)W[k * ncol + t * 16 + n] : (_Float16)0.0f;
}

// ---------------------------------------------------------------------------
// prep: h_init = lrelu(lrelu(x@W1+b1)@W2+b2).  One wave per 16-row tile.
// Stage 1: K=8 (padded to 32), Ncols=64 -> 4 WMMAs, result staged via LDS.
// Stage 2: K=64 (2 chunks), Ncols=32 -> 4 WMMAs.
__global__ void prep_kernel(const float* __restrict__ x,
                            const _Float16* __restrict__ W1f,  // frags nT=4,kc=1
                            const float* __restrict__ b1,      // [64]
                            const _Float16* __restrict__ W2f,  // frags nT=2,kc=2
                            const float* __restrict__ b2,      // [32]
                            float* __restrict__ h_init, int rows) {
    __shared__ _Float16 lds[8][16][64];
    int wave = threadIdx.x >> 5, lane = threadIdx.x & 31;
    int tile = blockIdx.x * 8 + wave;
    bool active = tile < (rows >> 4);         // uniform per wave -> EXEC all-1
    int c = lane >> 4, mn = lane & 15;

    if (active) {
        int ra = tile * 16 + mn;
        const float4* xr = (const float4*)(x + (size_t)ra * Ff);
        float4 p0 = xr[0], p1 = xr[1];        // x row (8 floats)
        bool lo = (c == 0);                   // K 0..7 valid only in low half
        v16h a;
        a[0] = (_Float16)(lo ? p0.x : 0.0f); a[1] = (_Float16)(lo ? p0.y : 0.0f);
        a[2] = (_Float16)(lo ? p0.z : 0.0f); a[3] = (_Float16)(lo ? p0.w : 0.0f);
        a[4] = (_Float16)(lo ? p1.x : 0.0f); a[5] = (_Float16)(lo ? p1.y : 0.0f);
        a[6] = (_Float16)(lo ? p1.z : 0.0f); a[7] = (_Float16)(lo ? p1.w : 0.0f);
#pragma unroll
        for (int e = 8; e < 16; ++e) a[e] = (_Float16)0.0f;
#pragma unroll
        for (int t = 0; t < 4; ++t) {
            v16h b = *(const v16h*)(W1f + ((size_t)t * 32 + lane) * 16);
            v8f acc = {};
            acc = wmma16(a, b, acc);
            int col = t * 16 + mn;
            float bv = b1[col];
#pragma unroll
            for (int j = 0; j < 8; ++j)
                lds[wave][j + 8 * c][col] = (_Float16)lrelu(acc[j] + bv);
        }
    }
    __syncthreads();
    if (active) {
        const v8h* lrow = (const v8h*)lds[wave][mn];   // 8 chunks of 8 halves
        v8h q0 = lrow[c], q1 = lrow[2 + c], q2 = lrow[4 + c], q3 = lrow[6 + c];
        v16h a0, a1;
#pragma unroll
        for (int e = 0; e < 8; ++e) {
            a0[e] = q0[e]; a0[8 + e] = q1[e];
            a1[e] = q2[e]; a1[8 + e] = q3[e];
        }
#pragma unroll
        for (int t = 0; t < 2; ++t) {
            v8f acc = {};
#pragma unroll
            for (int kc = 0; kc < 2; ++kc) {
                v16h b = *(const v16h*)(W2f + (((size_t)kc * 2 + t) * 32 + lane) * 16);
                acc = wmma16(kc ? a1 : a0, b, acc);
            }
            int col = t * 16 + mn;
            float bv = b2[col];
#pragma unroll
            for (int j = 0; j < 8; ++j) {
                int r = tile * 16 + j + 8 * c;
                h_init[r * DEMB + col] = lrelu(acc[j] + bv);
            }
        }
    }
}

// ---------------------------------------------------------------------------
// Generic [rows,32]@[32,32] GEMM + epilogue.  One wave per 16-row tile.
// mode 0: out = rmask[r] ? lrelu(A@B+bias) : 0            (init_h / msg)
// mode 2: out = rmask[r] ? addend + lrelu(A@B+bias) : fallback   (update)
// mode 3: out = lrelu(A@B+bias)                           (global encoder)
__global__ void gemm32_kernel(const float* __restrict__ A,
                              const _Float16* __restrict__ Bf,  // frags nT=2
                              const float* __restrict__ bias,   // [32]
                              const int* __restrict__ rmask,
                              const float* __restrict__ addend,
                              const float* __restrict__ fallback,
                              float* __restrict__ out,
                              int rows, int mode) {
    int wave = threadIdx.x >> 5, lane = threadIdx.x & 31;
    int tile = blockIdx.x * 8 + wave;
    if (tile >= (rows >> 4)) return;          // uniform per wave
    int c = lane >> 4, mn = lane & 15;
    int ra = tile * 16 + mn;
    const float4* Ar = (const float4*)(A + (size_t)ra * DEMB);
    v16h a = packA(Ar[2 * c], Ar[2 * c + 1], Ar[4 + 2 * c], Ar[5 + 2 * c]);
#pragma unroll
    for (int t = 0; t < 2; ++t) {
        v16h b = *(const v16h*)(Bf + ((size_t)t * 32 + lane) * 16);
        v8f acc = {};
        acc = wmma16(a, b, acc);
        int col = t * 16 + mn;
        float bv = bias[col];
#pragma unroll
        for (int j = 0; j < 8; ++j) {
            int r = tile * 16 + j + 8 * c;
            float v = lrelu(acc[j] + bv);
            float o;
            if (mode == 0)      o = rmask[r] ? v : 0.0f;
            else if (mode == 2) o = rmask[r] ? (addend[(size_t)r * DEMB + col] + v)
                                             : fallback[(size_t)r * DEMB + col];
            else                o = v;
            out[(size_t)r * DEMB + col] = o;
        }
    }
}

// ---------------------------------------------------------------------------
// DagEncoder GEMM: hcat = lrelu([x|h] @ Wdag + bdag), K=40 padded to 64.
__global__ void dag_kernel(const float* __restrict__ x,
                           const float* __restrict__ h,
                           const _Float16* __restrict__ Wf,  // frags nT=2,kc=2
                           const float* __restrict__ bias,   // [32]
                           float* __restrict__ hcat, int rows) {
    int wave = threadIdx.x >> 5, lane = threadIdx.x & 31;
    int tile = blockIdx.x * 8 + wave;
    if (tile >= (rows >> 4)) return;
    int c = lane >> 4, mn = lane & 15;
    int ra = tile * 16 + mn;
    const float4* hq = (const float4*)(h + (size_t)ra * DEMB);
    // a0 elements 8..15: K = 16+8c -> concat col 16+8c.. -> h cols 8+8c..15+8c
    float4 m0 = hq[2 + 2 * c], m1 = hq[3 + 2 * c];
    float4 l0, l1, u0, u1;
    if (c == 0) {
        const float4* xq = (const float4*)(x + (size_t)ra * Ff);
        l0 = xq[0]; l1 = xq[1];               // K 0..7   -> x cols 0..7
        u0 = hq[6]; u1 = hq[7];               // K 32..39 -> h cols 24..31
    } else {
        l0 = hq[0]; l1 = hq[1];               // K 8..15  -> h cols 0..7
        u0 = make_float4(0.f, 0.f, 0.f, 0.f); // K 40..47 -> pad
        u1 = u0;
    }
    v16h a0 = packA(l0, l1, m0, m1);
    float4 zz = make_float4(0.f, 0.f, 0.f, 0.f);
    v16h a1 = packA(u0, u1, zz, zz);          // K 48.. -> pad
#pragma unroll
    for (int t = 0; t < 2; ++t) {
        v8f acc = {};
#pragma unroll
        for (int kc = 0; kc < 2; ++kc) {
            v16h b = *(const v16h*)(Wf + (((size_t)kc * 2 + t) * 32 + lane) * 16);
            acc = wmma16(kc ? a1 : a0, b, acc);
        }
        int col = t * 16 + mn;
        float bv = bias[col];
#pragma unroll
        for (int j = 0; j < 8; ++j) {
            int r = tile * 16 + j + 8 * c;
            hcat[(size_t)r * DEMB + col] = lrelu(acc[j] + bv);
        }
    }
}

// ---------------------------------------------------------------------------
// Scatter / mask / zero helpers (pure bandwidth kernels).
__global__ void fill_int_kernel(int* __restrict__ p, int n, int v) {
    int tid = blockIdx.x * blockDim.x + threadIdx.x;
    if (tid < n) p[tid] = v;
}

__global__ void noin_scatter_kernel(const int* __restrict__ dst,
                                    int* __restrict__ noin, int ne) {
    int tid = blockIdx.x * blockDim.x + threadIdx.x;
    if (tid < ne) noin[dst[tid]] = 0;
}

// Zero agg (float4 stores) + masks.  n4 = Nn*8 float4s, rows = Nn.
__global__ void layer_zero_kernel(float4* __restrict__ agg4,
                                  int* __restrict__ smask,
                                  int* __restrict__ dmask, int n4, int rows) {
    int tid = blockIdx.x * blockDim.x + threadIdx.x;
    if (tid < n4) agg4[tid] = make_float4(0.f, 0.f, 0.f, 0.f);
    if (tid < rows) { smask[tid] = 0; dmask[tid] = 0; }
}

// edge_masks is a JAX bool array -> 1 byte per element.
__global__ void mask_scatter_kernel(const int* __restrict__ src,
                                    const int* __restrict__ dst,
                                    const unsigned char* __restrict__ em,
                                    int* __restrict__ smask,
                                    int* __restrict__ dmask, int ne) {
    int tid = blockIdx.x * blockDim.x + threadIdx.x;
    if (tid < ne && em[tid]) { smask[src[tid]] = 1; dmask[dst[tid]] = 1; }
}

// agg[dst[e]] += msg[src[e]]: thread per (edge, 4-channel group).
__global__ void scatter_add_kernel(const int* __restrict__ src,
                                   const int* __restrict__ dst,
                                   const unsigned char* __restrict__ em,
                                   const float* __restrict__ msg,
                                   float* __restrict__ agg, int ne) {
    int tid = blockIdx.x * blockDim.x + threadIdx.x;
    int e = tid >> 3, q = (tid & 7) * 4;
    if (e < ne && em[e]) {
        float4 mv = *(const float4*)(msg + (size_t)src[e] * DEMB + q);
        float* ap = agg + (size_t)dst[e] * DEMB + q;
        atomicAdd(ap + 0, mv.x);
        atomicAdd(ap + 1, mv.y);
        atomicAdd(ap + 2, mv.z);
        atomicAdd(ap + 3, mv.w);
    }
}

// Per-segment CSR sum: one wave per segment, lane = channel.
__global__ void segsum_kernel(const float* __restrict__ rowsrc,
                              const int* __restrict__ ptr,
                              float* __restrict__ out, int nseg) {
    int s = blockIdx.x;
    if (s >= nseg) return;
    int ch = threadIdx.x;                     // blockDim.x == 32
    int b = ptr[s], e = ptr[s + 1];
    float acc = 0.0f;
    for (int r = b; r < e; ++r) acc += rowsrc[(size_t)r * DEMB + ch];
    out[(size_t)s * DEMB + ch] = acc;
}

// ---------------------------------------------------------------------------
extern "C" void kernel_launch(void* const* d_in, const int* in_sizes, int n_in,
                              void* d_out, int out_size, void* d_ws, size_t ws_size,
                              hipStream_t stream) {
    (void)in_sizes; (void)n_in; (void)out_size; (void)ws_size;

    const float* x          = (const float*)d_in[0];
    const int*   edge_index = (const int*)d_in[1];
    const unsigned char* edge_masks = (const unsigned char*)d_in[2];
    const int*   ptr        = (const int*)d_in[3];
    const int*   obs_ptr    = (const int*)d_in[4];
    const float* w_prep1 = (const float*)d_in[5];  const float* b_prep1 = (const float*)d_in[6];
    const float* w_prep2 = (const float*)d_in[7];  const float* b_prep2 = (const float*)d_in[8];
    const float* w_msg   = (const float*)d_in[9];  const float* b_msg   = (const float*)d_in[10];
    const float* w_upd   = (const float*)d_in[11]; const float* b_upd   = (const float*)d_in[12];
    const float* w_dag   = (const float*)d_in[13]; const float* b_dag   = (const float*)d_in[14];
    const float* w_glob  = (const float*)d_in[15]; const float* b_glob  = (const float*)d_in[16];

    const int* dst = edge_index;        // edge_index[0]
    const int* src = edge_index + Ee;   // edge_index[1]

    // Outputs (flat concat): h [N,32], h_dag [4096,32], h_glob [128,32]
    float* h      = (float*)d_out;
    float* h_dag  = h + (size_t)Nn * DEMB;
    float* h_glob = h_dag + (size_t)Dseg * DEMB;

    // Workspace carve-out (256B aligned slices).
    char* ws = (char*)d_ws;
    size_t off = 0;
    auto carve = [&](size_t bytes) {
        void* p = ws + off;
        off = (off + bytes + 255) & ~(size_t)255;
        return p;
    };
    float* h_init = (float*)carve((size_t)Nn * DEMB * 4);
    float* msg    = (float*)carve((size_t)Nn * DEMB * 4);   // reused as hcat
    float* agg    = (float*)carve((size_t)Nn * DEMB * 4);
    int*   noin   = (int*)carve((size_t)Nn * 4);
    int*   smask  = (int*)carve((size_t)Nn * 4);
    int*   dmask  = (int*)carve((size_t)Nn * 4);
    float* h_d    = (float*)carve((size_t)Dseg * DEMB * 4);
    _Float16* W1f   = (_Float16*)carve(1 * 4 * 512 * 2);    // nkc=1, nT=4
    _Float16* W2f   = (_Float16*)carve(2 * 2 * 512 * 2);    // nkc=2, nT=2
    _Float16* Wmsgf = (_Float16*)carve(1 * 2 * 512 * 2);
    _Float16* Wupdf = (_Float16*)carve(1 * 2 * 512 * 2);
    _Float16* Wdagf = (_Float16*)carve(2 * 2 * 512 * 2);
    _Float16* Wglobf= (_Float16*)carve(1 * 2 * 512 * 2);

    const int TB = 256;
    // --- weight panels -> WMMA fragment order, f16, K-padded ---
    convert_frag_kernel<<<(4 * 512 + TB - 1) / TB, TB, 0, stream>>>(w_prep1, W1f, Ff, 1, Hh);
    convert_frag_kernel<<<(4 * 512 + TB - 1) / TB, TB, 0, stream>>>(w_prep2, W2f, Hh, 2, DEMB);
    convert_frag_kernel<<<(2 * 512 + TB - 1) / TB, TB, 0, stream>>>(w_msg, Wmsgf, DEMB, 1, DEMB);
    convert_frag_kernel<<<(2 * 512 + TB - 1) / TB, TB, 0, stream>>>(w_upd, Wupdf, DEMB, 1, DEMB);
    convert_frag_kernel<<<(4 * 512 + TB - 1) / TB, TB, 0, stream>>>(w_dag, Wdagf, Ff + DEMB, 2, DEMB);
    convert_frag_kernel<<<(2 * 512 + TB - 1) / TB, TB, 0, stream>>>(w_glob, Wglobf, DEMB, 1, DEMB);

    const int nTiles = Nn / 16;                    // 31250
    const int gB = (nTiles + 7) / 8;               // 8 waves/block

    // --- h_init = lrelu(lrelu(x@W1+b1)@W2+b2) ---
    prep_kernel<<<gB, TB, 0, stream>>>(x, W1f, b_prep1, W2f, b_prep2, h_init, Nn);

    // --- no_in mask ---
    fill_int_kernel<<<(Nn + TB - 1) / TB, TB, 0, stream>>>(noin, Nn, 1);
    noin_scatter_kernel<<<(Ee + TB - 1) / TB, TB, 0, stream>>>(dst, noin, Ee);

    // --- h = no_in ? lrelu(h_init@Wupd+b_upd) : 0 ---
    gemm32_kernel<<<gB, TB, 0, stream>>>(h_init, Wupdf, b_upd, noin,
                                         nullptr, nullptr, h, Nn, 0);

    // --- message-passing layers, l = L-1 .. 0 ---
    for (int l = Ll - 1; l >= 0; --l) {
        const unsigned char* em = edge_masks + (size_t)l * Ee;
        layer_zero_kernel<<<(Nn * 8 + TB - 1) / TB, TB, 0, stream>>>(
            (float4*)agg, smask, dmask, Nn * 8, Nn);
        mask_scatter_kernel<<<(Ee + TB - 1) / TB, TB, 0, stream>>>(
            src, dst, em, smask, dmask, Ee);
        gemm32_kernel<<<gB, TB, 0, stream>>>(h, Wmsgf, b_msg, smask,
                                             nullptr, nullptr, msg, Nn, 0);
        scatter_add_kernel<<<((size_t)Ee * 8 + TB - 1) / TB, TB, 0, stream>>>(
            src, dst, em, msg, agg, Ee);
        gemm32_kernel<<<gB, TB, 0, stream>>>(agg, Wupdf, b_upd, dmask,
                                             h_init, h, h, Nn, 2);
    }

    // --- DagEncoder: hcat = lrelu([x|h]@Wdag+b), segment_sum over ptr ---
    dag_kernel<<<gB, TB, 0, stream>>>(x, h, Wdagf, b_dag, msg /*hcat*/, Nn);
    segsum_kernel<<<Dseg, 32, 0, stream>>>(msg, ptr, h_dag, Dseg);

    // --- GlobalEncoder ---
    gemm32_kernel<<<(Dseg / 16 + 7) / 8, TB, 0, stream>>>(
        h_dag, Wglobf, b_glob, nullptr, nullptr, nullptr, h_d, Dseg, 3);
    segsum_kernel<<<Oseg, 32, 0, stream>>>(h_d, obs_ptr, h_glob, Oseg);
}
